// Qwen3MoeDecoderLayer_37426345017544
// MI455X (gfx1250) — matrix operand order repaired and checked by reading, not compiled, not run on hardware
//
#include <hip/hip_runtime.h>

// ---------------------------------------------------------------------------
// Qwen3-style MoE layer for MI455X (gfx1250, wave32, WMMA).
// Sparse top-8 routing + bf16 WMMA expert GEMMs (f32 accumulation).
// Double-buffered LDS pipeline; fragments fed by ds_load_b128; h-tiles staged
// with GLOBAL_LOAD_ASYNC_TO_LDS_B128 (s_wait_asynccnt fenced) when available.
// ---------------------------------------------------------------------------

typedef __attribute__((ext_vector_type(16))) __bf16 v16bf;
typedef __attribute__((ext_vector_type(8)))  __bf16 v8bf;
typedef __attribute__((ext_vector_type(8)))  float  v8f;
typedef __attribute__((ext_vector_type(4)))  int    v4i;

#define T_TOK 2048   // tokens (B*S)
#define HDIM  2048   // hidden
#define IDIM  768    // intermediate
#define NEXP  64     // experts
#define TOPK  8
#define NSLOT (T_TOK * TOPK)   // 16384 (token,slot) pairs
#define LDB   40     // padded LDS row stride in bf16 (80 B, 16-B aligned)

#if __has_builtin(__builtin_amdgcn_global_load_async_to_lds_b128) && \
    __has_builtin(__builtin_amdgcn_s_wait_asynccnt)
#define HAS_ASYNC_LDS 1
#endif
#define AS1 __attribute__((address_space(1)))
#define AS3 __attribute__((address_space(3)))

static __device__ __forceinline__ __bf16 cvt(float f) { return (__bf16)f; }

static __device__ __forceinline__ v8f wmma_bf16(v16bf a, v16bf b, v8f c) {
    return __builtin_amdgcn_wmma_f32_16x16x32_bf16(
        false, a, false, b, (short)0, c, false, false);
}

static __device__ __forceinline__ v16bf frag2(const __bf16* p0, const __bf16* p1) {
    v8bf lo = *(const v8bf*)p0;
    v8bf hi = *(const v8bf*)p1;
    return __builtin_shufflevector(lo, hi, 0, 1, 2, 3, 4, 5, 6, 7,
                                           8, 9, 10, 11, 12, 13, 14, 15);
}
// A fragment (16x32 MxK): lane%16 = M; two contiguous K runs per lane.
static __device__ __forceinline__ v16bf a_frag(const __bf16* aLds, int lane, int mBase) {
    const __bf16* p = aLds + (mBase + (lane & 15)) * LDB + ((lane & 16) ? 8 : 0);
    return frag2(p, p + 16);
}
// B fragment (32x16 KxN) from LDS tile stored transposed bT[n][k].
static __device__ __forceinline__ v16bf b_frag(const __bf16* bT, int lane, int nBase) {
    const __bf16* p = bT + (nBase + (lane & 15)) * LDB + ((lane & 16) ? 8 : 0);
    return frag2(p, p + 16);
}

// ---------------------------------------------------------------------------
__global__ void moe_zero(float* __restrict__ out, int* __restrict__ counts) {
    size_t i = (size_t)blockIdx.x * blockDim.x + threadIdx.x;
    if (i < (size_t)T_TOK * HDIM) out[i] = 0.f;
    if (i < NEXP) counts[i] = 0;
}

// ---------------------------------------------------------------------------
// Router logits = x @ Wr^T via WMMA.  Block = 32 tokens x 64 experts.
// ---------------------------------------------------------------------------
__global__ __launch_bounds__(256) void moe_router_logits(
        const float* __restrict__ x, const float* __restrict__ Wr,
        float* __restrict__ logits) {
    __shared__ __align__(16) __bf16 aLds[2][32 * LDB];
    __shared__ __align__(16) __bf16 bT[2][64 * LDB];
    const int tid = threadIdx.x, lane = tid & 31, w = tid >> 5;
    const int mi = w & 1, nj = w >> 1;
    const int t0 = blockIdx.x * 32;

    // staging: A = x tile (float4 over K), B = Wr tile ([E][H] -> float4 over K)
    const int ar = tid >> 3, ac4 = (tid & 7) * 4;            // 256 float4 groups
    const int bn = tid >> 3, bk4 = (tid & 7) * 4;            // x2 iterations
#define RT_STAGE(p, k0)                                                        \
    {                                                                          \
        float4 va = *(const float4*)&x[(size_t)(t0 + ar) * HDIM + (k0) + ac4]; \
        __bf16* d = &aLds[p][ar * LDB + ac4];                                  \
        d[0] = cvt(va.x); d[1] = cvt(va.y); d[2] = cvt(va.z); d[3] = cvt(va.w);\
        _Pragma("unroll")                                                      \
        for (int it = 0; it < 2; ++it) {                                       \
            int n = bn + it * 32;                                              \
            float4 vb = *(const float4*)&Wr[(size_t)n * HDIM + (k0) + bk4];    \
            __bf16* e = &bT[p][n * LDB + bk4];                                 \
            e[0] = cvt(vb.x); e[1] = cvt(vb.y); e[2] = cvt(vb.z); e[3] = cvt(vb.w); \
        }                                                                      \
    }
    v8f acc = {};
    RT_STAGE(0, 0);
    __syncthreads();
    for (int kt = 0; kt < HDIM / 32; ++kt) {
        const int p = kt & 1;
        if (kt + 1 < HDIM / 32) RT_STAGE(p ^ 1, (kt + 1) * 32);
        v16bf a = a_frag(aLds[p], lane, mi * 16);
        v16bf b = b_frag(bT[p], lane, nj * 16);
        acc = wmma_bf16(a, b, acc);
        __syncthreads();
    }
#undef RT_STAGE
    const int hi = (lane & 16) ? 8 : 0;
    const int gN = nj * 16 + (lane & 15);
#pragma unroll
    for (int r = 0; r < 8; ++r)
        logits[(size_t)(t0 + mi * 16 + r + hi) * NEXP + gN] = acc[r];
}

// ---------------------------------------------------------------------------
// Softmax + top-8 + renorm per token (one wave32); scatter to expert lists.
// ---------------------------------------------------------------------------
__global__ void moe_topk(const float* __restrict__ logits,
                         float* __restrict__ slotW,
                         int* __restrict__ counts, int* __restrict__ lists) {
    const int t = blockIdx.x, lane = threadIdx.x;
    float l0 = logits[(size_t)t * NEXP + lane];
    float l1 = logits[(size_t)t * NEXP + 32 + lane];
    float m = fmaxf(l0, l1);
    for (int off = 16; off; off >>= 1) m = fmaxf(m, __shfl_xor(m, off));
    float p0 = __expf(l0 - m), p1 = __expf(l1 - m);
    float z = p0 + p1;
    for (int off = 16; off; off >>= 1) z += __shfl_xor(z, off);
    p0 /= z; p1 /= z;
    float wv[TOPK]; int wi[TOPK]; float sum = 0.f;
#pragma unroll
    for (int k = 0; k < TOPK; ++k) {
        float v; int ix;
        if (p0 >= p1) { v = p0; ix = lane; } else { v = p1; ix = lane + 32; }
        for (int off = 16; off; off >>= 1) {
            float ov = __shfl_xor(v, off);
            int   oi = __shfl_xor(ix, off);
            if (ov > v || (ov == v && oi < ix)) { v = ov; ix = oi; }
        }
        wv[k] = v; wi[k] = ix; sum += v;
        if (ix == lane)      p0 = -1.f;
        if (ix == lane + 32) p1 = -1.f;
    }
    if (lane < TOPK) {
        int slot = t * TOPK + lane;
        slotW[slot] = wv[lane] / sum;
        int e = wi[lane];
        int pos = atomicAdd(&counts[e], 1);
        lists[(size_t)e * T_TOK + pos] = slot;
    }
}

// ---------------------------------------------------------------------------
// Gate/up GEMMs + fused SiLU; combine weight folded into h.
// Block = 64 gathered rows x 64 cols of I; 8 waves x 2 tiles (shared A frag).
// ---------------------------------------------------------------------------
__global__ __launch_bounds__(256) void moe_gateup(
        const float* __restrict__ x, const float* __restrict__ Wg,
        const float* __restrict__ Wu, const float* __restrict__ slotW,
        const int* __restrict__ counts, const int* __restrict__ lists,
        __bf16* __restrict__ hBuf) {
    const int e = blockIdx.z;
    const int cnt = counts[e];
    const int rBase = blockIdx.x * 64;
    if (rBase >= cnt) return;
    __shared__ __align__(16) __bf16 aLds[2][64 * LDB];
    __shared__ __align__(16) __bf16 bgT[2][64 * LDB];
    __shared__ __align__(16) __bf16 buT[2][64 * LDB];
    __shared__ int sSlot[64];
    const int tid = threadIdx.x, lane = tid & 31, w = tid >> 5;
    const int mi = w & 3;
    const int n0G = blockIdx.y * 64;
    if (tid < 64) {
        int r = rBase + tid;
        sSlot[tid] = (r < cnt) ? lists[(size_t)e * T_TOK + r] : -1;
    }
    __syncthreads();
    const float* wgB = Wg + (size_t)e * HDIM * IDIM;
    const float* wuB = Wu + (size_t)e * HDIM * IDIM;

    const int ar = tid >> 3, ac4 = (tid & 7) * 4;            // A: 64x32, 2 iters
    const int bk = tid >> 4, bn4 = (tid & 15) * 4;           // B: 32x64, 2 iters
#define GU_STAGE(p, k0)                                                        \
    {                                                                          \
        _Pragma("unroll")                                                      \
        for (int it = 0; it < 2; ++it) {                                       \
            int r = ar + it * 32;                                              \
            int slot = sSlot[r];                                               \
            float4 va = make_float4(0.f, 0.f, 0.f, 0.f);                       \
            if (slot >= 0)                                                     \
                va = *(const float4*)&x[(size_t)(slot >> 3) * HDIM + (k0) + ac4]; \
            __bf16* d = &aLds[p][r * LDB + ac4];                               \
            d[0] = cvt(va.x); d[1] = cvt(va.y); d[2] = cvt(va.z); d[3] = cvt(va.w); \
        }                                                                      \
        _Pragma("unroll")                                                      \
        for (int it = 0; it < 2; ++it) {                                       \
            int k = bk + it * 16;                                              \
            size_t off = (size_t)((k0) + k) * IDIM + n0G + bn4;                \
            float4 vg = *(const float4*)&wgB[off];                             \
            float4 vu = *(const float4*)&wuB[off];                             \
            _Pragma("unroll")                                                  \
            for (int j = 0; j < 4; ++j) {                                      \
                bgT[p][(bn4 + j) * LDB + k] = cvt(((const float*)&vg)[j]);     \
                buT[p][(bn4 + j) * LDB + k] = cvt(((const float*)&vu)[j]);     \
            }                                                                  \
        }                                                                      \
    }
    v8f cg[2] = {{}, {}}, cu[2] = {{}, {}};
    GU_STAGE(0, 0);
    __syncthreads();
    for (int kt = 0; kt < HDIM / 32; ++kt) {
        const int p = kt & 1;
        if (kt + 1 < HDIM / 32) {
            GU_STAGE(p ^ 1, (kt + 1) * 32);
            __builtin_prefetch(wgB + (size_t)((kt + 2) * 32 + (tid >> 3)) * IDIM + n0G, 0, 1);
            __builtin_prefetch(wuB + (size_t)((kt + 2) * 32 + (tid >> 3)) * IDIM + n0G, 0, 1);
        }
        v16bf a = a_frag(aLds[p], lane, mi * 16);
#pragma unroll
        for (int t = 0; t < 2; ++t) {
            int nj = (w >> 2) + t * 2;
            v16bf bg = b_frag(bgT[p], lane, nj * 16);
            v16bf bu = b_frag(buT[p], lane, nj * 16);
            cg[t] = wmma_bf16(a, bg, cg[t]);
            cu[t] = wmma_bf16(a, bu, cu[t]);
        }
        __syncthreads();
    }
#undef GU_STAGE
    const int hi = (lane & 16) ? 8 : 0;
#pragma unroll
    for (int t = 0; t < 2; ++t) {
        int gN = n0G + ((w >> 2) + t * 2) * 16 + (lane & 15);
#pragma unroll
        for (int r = 0; r < 8; ++r) {
            int row = mi * 16 + r + hi;
            int slot = sSlot[row];
            if (slot >= 0) {
                float g = cg[t][r], u = cu[t][r];
                float h = (g / (1.f + __expf(-g))) * u * slotW[slot];
                hBuf[(size_t)slot * IDIM + gN] = cvt(h);
            }
        }
    }
}

// ---------------------------------------------------------------------------
// Down-proj (w*h) @ Wd, scatter-accumulate into out via f32 atomics.
// h tiles are raw bf16 -> staged with GLOBAL_LOAD_ASYNC_TO_LDS_B128 when the
// toolchain exposes it (s_wait_asynccnt fences before each barrier).
// ---------------------------------------------------------------------------
__global__ __launch_bounds__(256) void moe_down(
        const __bf16* __restrict__ hBuf, const float* __restrict__ Wd,
        const int* __restrict__ counts, const int* __restrict__ lists,
        float* __restrict__ out) {
    const int e = blockIdx.z;
    const int cnt = counts[e];
    const int rBase = blockIdx.x * 64;
    if (rBase >= cnt) return;
    __shared__ __align__(16) __bf16 aLds[2][64 * LDB];
    __shared__ __align__(16) __bf16 bT[2][64 * LDB];
    __shared__ int sSlot[64];
    const int tid = threadIdx.x, lane = tid & 31, w = tid >> 5;
    const int mi = w & 3;
    const int n0G = blockIdx.y * 64;
    if (tid < 64) {
        int r = rBase + tid;
        sSlot[tid] = (r < cnt) ? lists[(size_t)e * T_TOK + r] : -1;
    }
    __syncthreads();
    const float* wdB = Wd + (size_t)e * IDIM * HDIM;

    const int ar = tid >> 2, as8 = (tid & 3) * 8;            // A: 64 rows x 4 segs
    const int bk = tid >> 4, bn4 = (tid & 15) * 4;           // B: 32x64, 2 iters
#ifdef HAS_ASYNC_LDS
    // builtin signature (from hipcc diagnostics): (v4i* global, v4i* lds, Ii, Ii)
#define HCOPY16(l, g) __builtin_amdgcn_global_load_async_to_lds_b128(          \
        (AS1 v4i*)(g), (AS3 v4i*)(l), 0, 0)
#define HWAIT() __builtin_amdgcn_s_wait_asynccnt(0)
#else
#define HCOPY16(l, g) (*(v8bf*)(l) = *(const v8bf*)(g))
#define HWAIT() ((void)0)
#endif
#define DN_STAGE(p, k0)                                                        \
    {                                                                          \
        int slot = sSlot[ar];                                                  \
        __bf16* l = &aLds[p][ar * LDB + as8];                                  \
        if (slot >= 0) {                                                       \
            const __bf16* g = &hBuf[(size_t)slot * IDIM + (k0) + as8];         \
            HCOPY16(l, g);                                                     \
        } else {                                                               \
            v8bf zz = {};                                                      \
            *(v8bf*)l = zz;                                                    \
        }                                                                      \
        _Pragma("unroll")                                                      \
        for (int it = 0; it < 2; ++it) {                                       \
            int k = bk + it * 16;                                              \
            float4 vd = *(const float4*)&wdB[(size_t)((k0) + k) * HDIM + n0G + bn4]; \
            _Pragma("unroll")                                                  \
            for (int j = 0; j < 4; ++j)                                        \
                bT[p][(bn4 + j) * LDB + k] = cvt(((const float*)&vd)[j]);      \
        }                                                                      \
    }
    v8f acc[2] = {{}, {}};
    DN_STAGE(0, 0);
    HWAIT();
    __syncthreads();
    for (int kt = 0; kt < IDIM / 32; ++kt) {
        const int p = kt & 1;
        if (kt + 1 < IDIM / 32) {
            DN_STAGE(p ^ 1, (kt + 1) * 32);
            __builtin_prefetch(wdB + (size_t)((kt + 2) * 32 + (tid >> 3)) * HDIM + n0G, 0, 1);
        }
        v16bf a = a_frag(aLds[p], lane, mi * 16);
#pragma unroll
        for (int t = 0; t < 2; ++t) {
            int nj = (w >> 2) + t * 2;
            v16bf b = b_frag(bT[p], lane, nj * 16);
            acc[t] = wmma_bf16(a, b, acc[t]);
        }
        HWAIT();
        __syncthreads();
    }
#undef DN_STAGE
#undef HCOPY16
#undef HWAIT
    const int hi = (lane & 16) ? 8 : 0;
#pragma unroll
    for (int t = 0; t < 2; ++t) {
        int gN = n0G + ((w >> 2) + t * 2) * 16 + (lane & 15);
#pragma unroll
        for (int r = 0; r < 8; ++r) {
            int row = mi * 16 + r + hi;
            int slot = sSlot[row];
            if (slot >= 0)
                atomicAdd(&out[(size_t)(slot >> 3) * HDIM + gN], acc[t][r]);
        }
    }
}

// ---------------------------------------------------------------------------
// Launcher. Workspace (~26.3 MB): logits | slotW | counts | lists | hBuf.
// ---------------------------------------------------------------------------
extern "C" void kernel_launch(void* const* d_in, const int* in_sizes, int n_in,
                              void* d_out, int out_size, void* d_ws, size_t ws_size,
                              hipStream_t stream) {
    (void)in_sizes; (void)n_in; (void)out_size; (void)ws_size;
    const float* x  = (const float*)d_in[0];
    const float* Wr = (const float*)d_in[1];
    const float* Wg = (const float*)d_in[2];
    const float* Wu = (const float*)d_in[3];
    const float* Wd = (const float*)d_in[4];
    float* out = (float*)d_out;

    char* ws = (char*)d_ws;
    size_t off = 0;
    float* logits = (float*)(ws + off); off += (size_t)T_TOK * NEXP * sizeof(float);
    float* slotW  = (float*)(ws + off); off += (size_t)NSLOT * sizeof(float);
    int*   counts = (int*)(ws + off);   off += 256 * sizeof(int);
    int*   lists  = (int*)(ws + off);   off += (size_t)NEXP * T_TOK * sizeof(int);
    __bf16* hBuf  = (__bf16*)(ws + off); off += (size_t)NSLOT * IDIM * sizeof(__bf16);

    moe_zero<<<((size_t)T_TOK * HDIM + 255) / 256, 256, 0, stream>>>(out, counts);
    moe_router_logits<<<T_TOK / 32, 256, 0, stream>>>(x, Wr, logits);
    moe_topk<<<T_TOK, 32, 0, stream>>>(logits, slotW, counts, lists);
    moe_gateup<<<dim3(32, IDIM / 64, NEXP), 256, 0, stream>>>(
        x, Wg, Wu, slotW, counts, lists, hBuf);
    moe_down<<<dim3(32, HDIM / 64, NEXP), 256, 0, stream>>>(
        hBuf, Wd, counts, lists, out);
}
